// CPT_AS_88261577933023
// MI455X (gfx1250) — matrix operand order repaired and checked by reading, not compiled, not run on hardware
//
#include <hip/hip_runtime.h>
#include <hip/hip_bf16.h>

// Problem constants (from reference): B=8, L1=L2=2048, D=1024
#define BQ  8
#define L1N 2048
#define L2N 2048
#define DM  1024
#define BL  (BQ * L1N) // 16384 total query rows

typedef __bf16 bf16_t;
typedef bf16_t v4bf  __attribute__((ext_vector_type(4)));
typedef bf16_t v8bf  __attribute__((ext_vector_type(8)));
typedef bf16_t v16bf __attribute__((ext_vector_type(16)));
typedef float  v8f   __attribute__((ext_vector_type(8)));

// D = A(16x32 bf16) * B(32x16 bf16) + C(16x16 f32), wave32
__device__ __forceinline__ v8f wmma_bf16(v16bf a, v16bf b, v8f c) {
  return __builtin_amdgcn_wmma_f32_16x16x32_bf16(
      /*neg_a=*/false, a, /*neg_b=*/false, b,
      /*c_mod=*/(short)0, c, /*reuse_a=*/false, /*reuse_b=*/false);
}

// A-fragment from row-major bf16 (16x32 A, contiguous k order, same order as the B operands):
// lane m = lane&15 holds K = {16*half..16*half+15}
// caller passes p = &A[m][16*half]; two contiguous 16B chunks at +0 and +8.
__device__ __forceinline__ v16bf load_a_frag(const bf16_t* p) {
  v8bf lo = *(const v8bf*)(p);
  v8bf hi = *(const v8bf*)(p + 8);
  return __builtin_shufflevector(lo, hi, 0,1,2,3,4,5,6,7,8,9,10,11,12,13,14,15);
}

// ---- CDNA5 async global->LDS copy (ASYNCcnt path) ----
// LDS flat-aperture addresses carry the LDS byte offset in addr[31:0].
__device__ __forceinline__ unsigned int lds_off_u32(const void* p) {
  return (unsigned int)(uintptr_t)p;
}
__device__ __forceinline__ void async_g2l_b128(const void* gsrc, const void* ldst) {
  asm volatile("global_load_async_to_lds_b128 %0, %1, off"
               :: "v"(lds_off_u32(ldst)),
                  "v"((unsigned long long)(uintptr_t)gsrc)
               : "memory");
}
__device__ __forceinline__ void wait_async0() {
  asm volatile("s_wait_asynccnt 0" ::: "memory");
}

// ---------------- pack / convert kernels ----------------

__global__ void k_f32_to_bf16(const float* __restrict__ s, bf16_t* __restrict__ d, int n4) {
  int i = blockIdx.x * blockDim.x + threadIdx.x;
  if (i >= n4) return;
  const float4 v = ((const float4*)s)[i];
  v4bf o; o[0] = (bf16_t)v.x; o[1] = (bf16_t)v.y; o[2] = (bf16_t)v.z; o[3] = (bf16_t)v.w;
  ((v4bf*)d)[i] = o;
}

// Pack a row-major f32 [K x N] matrix into bf16 B-fragment-linear layout:
// frag f = (k0/32)*(N/16) + n0/16 ; dst[(f*32 + lane)*16 + j] = src[k0+16*(lane>>4)+j][n0+(lane&15)]
__global__ void k_pack_b(const float* __restrict__ src, bf16_t* __restrict__ dst,
                         int K, int N, int ld,
                         long long srcBatchStride, long long dstBatchStride) {
  src += (long long)blockIdx.y * srcBatchStride;
  dst += (long long)blockIdx.y * dstBatchStride;
  int tid = blockIdx.x * blockDim.x + threadIdx.x;
  int total = (K / 32) * (N / 16) * 32;
  if (tid >= total) return;
  int lane = tid & 31, frag = tid >> 5;
  int nt = frag % (N / 16), kt = frag / (N / 16);
  int k = kt * 32 + ((lane >> 4) << 4);
  int n = nt * 16 + (lane & 15);
  const float* sp = src + (size_t)k * ld + n;
  bf16_t* dp = dst + (size_t)tid * 16;
#pragma unroll
  for (int j = 0; j < 16; ++j) dp[j] = (bf16_t)sp[(size_t)j * ld];
}

// ---------------- attention: scores -> softmax -> alpha @ ht ----------------
// One workgroup (8 wave32) per 16-row block.
// LDS: 16x2048 f32 scores (128KB) + 16x2048 bf16 alpha (64KB)
//      + 16x1024 bf16 h-tile (32KB, async-staged) = 224KB dynamic (<=320KB/WGP).

__global__ void __launch_bounds__(256) k_attn(
    const bf16_t* __restrict__ h_bf, const bf16_t* __restrict__ ht_bf,
    const bf16_t* __restrict__ htv, bf16_t* __restrict__ r_bf) {
  extern __shared__ char smem[];
  float*  s_sc = (float*)smem;                                             // [16][2048] f32
  bf16_t* s_al = (bf16_t*)(smem + (size_t)16 * L2N * sizeof(float));       // [16][2048] bf16
  bf16_t* s_h  = (bf16_t*)(smem + (size_t)16 * L2N * (sizeof(float) + 2)); // [16][1024] bf16

  const int tid  = threadIdx.x;
  const int lane = tid & 31, wave = tid >> 5;
  const int hlf  = lane >> 4, l15 = lane & 15;
  const int row0 = blockIdx.x * 16;         // global query row
  const int bat  = row0 >> 11;              // row0 / L1
  const bf16_t* htb  = ht_bf + (size_t)bat * L2N * DM;
  const bf16_t* htvb = htv   + (size_t)bat * L2N * DM;

  // Stage the 16xD h tile into LDS once (each wave reuses it for 16 column tiles).
  {
    const bf16_t* g = h_bf + (size_t)row0 * DM;   // contiguous 32KB
    for (int c = tid; c < (16 * DM) / 8; c += 256)  // 2048 chunks of 16B
      async_g2l_b128(g + (size_t)c * 8, s_h + (size_t)c * 8);
    wait_async0();
  }
  __syncthreads();

  // Phase 1: S = h @ ht^T  (A from LDS h-tile; B[k=d][n=key]=ht[key][d] contiguous per lane)
  const bf16_t* arow = s_h + (size_t)l15 * DM + 16 * hlf;
  for (int t = wave; t < L2N / 16; t += 8) {
    const int col0 = t * 16;
    const bf16_t* brow = htb + (size_t)(col0 + l15) * DM + 16 * hlf;
    v8f acc = {};
    for (int k0 = 0; k0 < DM; k0 += 32) {
      v16bf a  = load_a_frag(arow + k0);
      v16bf bb = *(const v16bf*)(brow + k0);
      acc = wmma_bf16(a, bb, acc);
    }
#pragma unroll
    for (int rr = 0; rr < 8; ++rr)
      s_sc[(size_t)(rr + 8 * hlf) * L2N + col0 + l15] = acc[rr];
  }
  __syncthreads();

  // Phase 2: row softmax; 16 threads per row (same 16-lane half-wave)
  {
    const int row = tid >> 4, sub = tid & 15;
    float* sr = s_sc + (size_t)row * L2N;
    float m = -1e30f;
    for (int c = sub; c < L2N; c += 16) m = fmaxf(m, sr[c]);
#pragma unroll
    for (int off = 8; off >= 1; off >>= 1) m = fmaxf(m, __shfl_xor(m, off, 32));
    float sum = 0.f;
    for (int c = sub; c < L2N; c += 16) { float e = __expf(sr[c] - m); sr[c] = e; sum += e; }
#pragma unroll
    for (int off = 8; off >= 1; off >>= 1) sum += __shfl_xor(sum, off, 32);
    const float inv = 1.f / sum;
    bf16_t* ar = s_al + (size_t)row * L2N;
    for (int c = sub; c < L2N; c += 16) ar[c] = (bf16_t)(sr[c] * inv);
  }
  __syncthreads();

  // Phase 3: r = alpha @ ht (A from LDS alpha, B from packed V fragments)
  const bf16_t* ap = s_al + (size_t)l15 * L2N + 16 * hlf;
  for (int t = wave; t < DM / 16; t += 8) {
    const int n0 = t * 16;
    v8f acc = {};
    for (int k0 = 0; k0 < L2N; k0 += 32) {
      v16bf a = load_a_frag(ap + k0);
      const size_t fo = (((size_t)(k0 >> 5) * (DM / 16) + t) * 32 + lane) * 16;
      v16bf bb = *(const v16bf*)(htvb + fo);
      acc = wmma_bf16(a, bb, acc);
    }
#pragma unroll
    for (int rr = 0; rr < 8; ++rr)
      r_bf[(size_t)(row0 + rr + 8 * hlf) * DM + n0 + l15] = (bf16_t)acc[rr];
  }
}

// ---------------- fused output: gate + 2-GEMM + tanh/sigmoid combine ----------------
// All 8 waves of a block share one weight-column strip (tileCol); the three
// 32KB weight strips are async-staged into LDS once per block (8x reuse).

__global__ void __launch_bounds__(256) k_out(
    const float* __restrict__ h, const float* __restrict__ pw,
    const bf16_t* __restrict__ h_bf, const bf16_t* __restrict__ r_bf,
    const bf16_t* __restrict__ wg, const bf16_t* __restrict__ w1,
    const bf16_t* __restrict__ w2,
    const float* __restrict__ b_gate, const float* __restrict__ b_lin,
    float* __restrict__ out) {
  extern __shared__ char smem[];
  bf16_t* s_w = (bf16_t*)smem;  // [3][32 kfrag][32 lane][16] bf16 = 96KB

  const int tid  = threadIdx.x;
  const int lane = tid & 31, wave = tid >> 5;
  const int hlf = lane >> 4, l15 = lane & 15;
  const int tileCol = blockIdx.x & 63;               // shared by the whole block
  const int tileRow = (blockIdx.x >> 6) * 8 + wave;  // one row tile per wave
  const int row0 = tileRow * 16, col0 = tileCol * 16;

  // Async-stage the three weight fragment strips for this column (3 x 32KB).
  for (int c = tid; c < 3 * 32 * 32; c += 256) {     // 3072 chunks of 32B
    const int w = c >> 10, rem = c & 1023;
    const int k = rem >> 5, ln = rem & 31;
    const bf16_t* wp = (w == 0) ? wg : ((w == 1) ? w1 : w2);
    const bf16_t* g = wp + (((size_t)k * 64 + tileCol) * 32 + ln) * 16;
    bf16_t* l = s_w + (size_t)c * 16;
    async_g2l_b128(g, l);
    async_g2l_b128(g + 8, l + 8);
  }
  wait_async0();
  __syncthreads();

  v8f accg = {}, accu = {};
  const bf16_t* ah = h_bf + (size_t)(row0 + l15) * DM + 16 * hlf;
  const bf16_t* ar = r_bf + (size_t)(row0 + l15) * DM + 16 * hlf;
  for (int k0 = 0; k0 < DM; k0 += 32) {
    if (k0 + 128 < DM) {  // prefetch A streams ahead (global_prefetch_b8)
      __builtin_prefetch(ah + k0 + 128, 0, 0);
      __builtin_prefetch(ar + k0 + 128, 0, 0);
    }
    v16bf a_h = load_a_frag(ah + k0);
    v16bf a_r = load_a_frag(ar + k0);
    const bf16_t* lb = s_w + (((size_t)(k0 >> 5) * 32) + lane) * 16;
    v16bf bg = *(const v16bf*)(lb);
    v16bf b1 = *(const v16bf*)(lb + 32 * 32 * 16);
    v16bf b2 = *(const v16bf*)(lb + 2 * 32 * 32 * 16);
    accg = wmma_bf16(a_h, bg, accg);   // h @ W_gate
    accu = wmma_bf16(a_r, b1, accu);   // r @ W_lin[:D]
    accu = wmma_bf16(a_h, b2, accu);   // h @ W_lin[D:]
  }

  const int col = col0 + l15;
  const float bgv = b_gate[col], blv = b_lin[col];
#pragma unroll
  for (int rr = 0; rr < 8; ++rr) {
    const int row = row0 + rr + 8 * hlf;
    const float tgate = 1.f / (1.f + __expf(-(accg[rr] + bgv)));
    const float hn = tanhf(accu[rr] + blv) * pw[row];
    const float hv = h[(size_t)row * DM + col];
    out[(size_t)row * DM + col] = tgate * hn + (1.f - tgate) * hv;
  }
}

// ---------------- host launcher ----------------

extern "C" void kernel_launch(void* const* d_in, const int* in_sizes, int n_in,
                              void* d_out, int out_size, void* d_ws, size_t ws_size,
                              hipStream_t stream) {
  const float* h  = (const float*)d_in[0];   // [B,L1,D]
  const float* ht = (const float*)d_in[1];   // [B,L2,D]
  const float* pw = (const float*)d_in[2];   // [B,L1]
  const float* Wg = (const float*)d_in[3];   // [D,D]
  const float* bg = (const float*)d_in[4];   // [D]
  const float* Wl = (const float*)d_in[5];   // [2D,D]
  const float* bl = (const float*)d_in[6];   // [D]
  float* out = (float*)d_out;

  // workspace layout (~134 MB total)
  char* ws = (char*)d_ws;
  bf16_t* h_bf  = (bf16_t*)ws;                 ws += (size_t)BL * DM * 2;        // 32MB
  bf16_t* ht_bf = (bf16_t*)ws;                 ws += (size_t)BQ * L2N * DM * 2;  // 32MB
  bf16_t* htv   = (bf16_t*)ws;                 ws += (size_t)BQ * L2N * DM * 2;  // 32MB (V packed)
  bf16_t* wgp   = (bf16_t*)ws;                 ws += (size_t)DM * DM * 2;        // 2MB
  bf16_t* w1p   = (bf16_t*)ws;                 ws += (size_t)DM * DM * 2;        // 2MB
  bf16_t* w2p   = (bf16_t*)ws;                 ws += (size_t)DM * DM * 2;        // 2MB
  bf16_t* r_bf  = (bf16_t*)ws;                 ws += (size_t)BL * DM * 2;        // 32MB

  // 1) f32 -> bf16 conversions (vectorized x4)
  {
    int n4 = BL * DM / 4;
    k_f32_to_bf16<<<(n4 + 255) / 256, 256, 0, stream>>>(h, h_bf, n4);
    n4 = BQ * L2N * DM / 4;
    k_f32_to_bf16<<<(n4 + 255) / 256, 256, 0, stream>>>(ht, ht_bf, n4);
  }

  // 2) pack B operands into per-lane fragment-linear layout
  {
    const int wthreads = (DM / 32) * (DM / 16) * 32;  // 65536
    k_pack_b<<<dim3(wthreads / 256), 256, 0, stream>>>(Wg, wgp, DM, DM, DM, 0, 0);
    k_pack_b<<<dim3(wthreads / 256), 256, 0, stream>>>(Wl, w1p, DM, DM, DM, 0, 0);
    k_pack_b<<<dim3(wthreads / 256), 256, 0, stream>>>(Wl + (size_t)DM * DM, w2p, DM, DM, DM, 0, 0);
    const int vthreads = (L2N / 32) * (DM / 16) * 32; // 131072 per batch
    k_pack_b<<<dim3(vthreads / 256, BQ), 256, 0, stream>>>(
        ht, htv, L2N, DM, DM, (long long)L2N * DM, (long long)L2N * DM);
  }

  // 3) attention (scores + softmax + alpha@ht), 224KB dynamic LDS
  {
    const size_t lds = (size_t)16 * L2N * sizeof(float)   // scores f32
                     + (size_t)16 * L2N * sizeof(bf16_t)  // alpha bf16
                     + (size_t)16 * DM  * sizeof(bf16_t); // h tile bf16
    k_attn<<<BL / 16, 256, lds, stream>>>(h_bf, ht_bf, htv, r_bf);
  }

  // 4) fused gate + output GEMMs + combine, 96KB dynamic LDS
  {
    const int tiles = (BL / 16) * (DM / 16);  // 65536 tiles, 8 waves/block
    const size_t lds = (size_t)3 * 32 * 32 * 16 * sizeof(bf16_t);
    k_out<<<tiles / 8, 256, lds, stream>>>(h, pw, h_bf, r_bf, wgp, w1p, w2p, bg, bl, out);
  }
}
